// NodeModel_18786186952980
// MI455X (gfx1250) — compile-verified, matrix operand order checked by reading, not run on hardware
//
#include <hip/hip_runtime.h>

#define BDIM 2
#define NN 50000
#define EE 300000
#define NTILES (NN / 16)   // 3125

typedef __bf16 v16bf __attribute__((ext_vector_type(16)));
typedef __bf16 v8bf  __attribute__((ext_vector_type(8)));
typedef float  v8f   __attribute__((ext_vector_type(8)));

union BFrag { v16bf v; v8bf h[2]; };

// ---------------- workspace zero (sums + counts, contiguous) ----------------
__global__ __launch_bounds__(256) void zero_ws(float* __restrict__ p, long n) {
    long i = (long)blockIdx.x * 256 + threadIdx.x;
    if (i < n) p[i] = 0.0f;
}

// ---------------- W1/W2 -> bf16, transposed to [N][K] (K contiguous) --------
__global__ __launch_bounds__(256) void convert_weights(const float* __restrict__ W1,
                                                       const float* __restrict__ W2,
                                                       __bf16* __restrict__ W1T,
                                                       __bf16* __restrict__ W2T) {
    int i = blockIdx.x * 256 + threadIdx.x;          // 0 .. 32767
    if (i < 256 * 128) {
        int k = i >> 7, h = i & 127;                 // W1: [256][128] row-major
        W1T[h * 256 + k] = (__bf16)W1[i];
    }
    if (i < 128 * 128) {
        int k = i >> 7, h = i & 127;                 // W2: [128][128] row-major
        W2T[h * 128 + k] = (__bf16)W2[i];
    }
}

// ---------------- scatter-add edges into per-node sums + counts -------------
__global__ __launch_bounds__(256) void scatter_edges(const float* __restrict__ edges,
                                                     const int* __restrict__ recv,
                                                     float* __restrict__ sums,
                                                     float* __restrict__ cnt) {
    long idx = (long)blockIdx.x * 256 + threadIdx.x; // < B*E*128
    long be  = idx >> 7;
    int  d   = (int)(idx & 127);
    int  b   = (int)(be / EE);
    int  r   = recv[be];
    long dst = ((long)b * NN + r) * 128 + d;
    unsafeAtomicAdd(&sums[dst], edges[idx]);
    if (d == 0) unsafeAtomicAdd(&cnt[(long)b * NN + r], 1.0f);
}

// ---------------- fused: concat -> GEMM1 -> ReLU -> GEMM2 -> LN -> residual -
__global__ __launch_bounds__(256) void node_mlp(const float* __restrict__ nf,
                                                const float* __restrict__ sums,
                                                const float* __restrict__ cnt,
                                                const __bf16* __restrict__ W1T,
                                                const __bf16* __restrict__ W2T,
                                                const float* __restrict__ b1,
                                                const float* __restrict__ b2,
                                                const float* __restrict__ lns,
                                                const float* __restrict__ lnb,
                                                float* __restrict__ out) {
    __shared__ __bf16 featA[16][264];   // padded: row stride 528B (16B aligned)
    __shared__ __bf16 hA[16][136];      // padded: row stride 272B
    __shared__ float  outS[16][128];
    __shared__ float  muS[16], rsS[16];

    const int  tid  = threadIdx.x;
    const int  tile = blockIdx.x;
    const int  b    = tile / NTILES;
    const long row0 = (long)(tile % NTILES) * 16;
    const long base = ((long)b * NN + row0) * 128;

    // Stage A-matrix: feat[16][256] = [node_features | scatter-mean], bf16.
    #pragma unroll
    for (int i = 0; i < 16; ++i) {
        int e = tid + i * 256;          // 0..4095
        int m = e >> 8;
        int c = e & 255;
        float v;
        if (c < 128) {
            v = nf[base + (long)m * 128 + c];
        } else {
            float cn = cnt[(long)b * NN + row0 + m];
            v = sums[base + (long)m * 128 + (c - 128)] / fmaxf(cn, 1.0f);
        }
        featA[m][c] = (__bf16)v;
    }
    __syncthreads();

    const int wave = tid >> 5;
    const int lane = tid & 31;
    const int ll   = lane & 15;   // A: row M ; B/C: column N
    const int lh   = lane >> 4;   // K half-select / M high bit
    const int n0   = wave * 16;   // this wave's 16-wide column slab

    // ---- GEMM1: feat[16,256] x W1[256, n0:n0+16]  (8 x wmma 16x16x32 bf16)
    v8f acc = {};
    #pragma unroll
    for (int kk = 0; kk < 8; ++kk) {
        int kb = kk * 32 + lh * 8;
        BFrag a, w;
        a.h[0] = *(const v8bf*)&featA[ll][kb];
        a.h[1] = *(const v8bf*)&featA[ll][kb + 16];
        const __bf16* wp = &W1T[(n0 + ll) * 256 + kb];
        w.h[0] = *(const v8bf*)wp;
        w.h[1] = *(const v8bf*)(wp + 16);
        acc = __builtin_amdgcn_wmma_f32_16x16x32_bf16(false, a.v, false, w.v,
                                                      (short)0, acc, false, false);
    }
    {
        float bias = b1[n0 + ll];
        #pragma unroll
        for (int r = 0; r < 8; ++r) {
            float v = acc[r] + bias;
            v = v > 0.0f ? v : 0.0f;                 // ReLU
            hA[r + 8 * lh][n0 + ll] = (__bf16)v;     // M = r + 8*lh, N = n0+ll
        }
    }
    __syncthreads();

    // ---- GEMM2: h[16,128] x W2[128, n0:n0+16]  (4 x wmma 16x16x32 bf16)
    v8f acc2 = {};
    #pragma unroll
    for (int kk = 0; kk < 4; ++kk) {
        int kb = kk * 32 + lh * 8;
        BFrag a, w;
        a.h[0] = *(const v8bf*)&hA[ll][kb];
        a.h[1] = *(const v8bf*)&hA[ll][kb + 16];
        const __bf16* wp = &W2T[(n0 + ll) * 128 + kb];
        w.h[0] = *(const v8bf*)wp;
        w.h[1] = *(const v8bf*)(wp + 16);
        acc2 = __builtin_amdgcn_wmma_f32_16x16x32_bf16(false, a.v, false, w.v,
                                                       (short)0, acc2, false, false);
    }
    {
        float bias = b2[n0 + ll];
        #pragma unroll
        for (int r = 0; r < 8; ++r)
            outS[r + 8 * lh][n0 + ll] = acc2[r] + bias;
    }
    __syncthreads();

    // ---- LayerNorm stats (one thread per row; 16*128 LDS reads, negligible)
    if (tid < 16) {
        float s = 0.0f, s2 = 0.0f;
        #pragma unroll 4
        for (int c = 0; c < 128; ++c) {
            float v = outS[tid][c];
            s += v; s2 += v * v;
        }
        float mu  = s * (1.0f / 128.0f);
        float var = s2 * (1.0f / 128.0f) - mu * mu;
        muS[tid] = mu;
        rsS[tid] = rsqrtf(var + 1e-5f);
    }
    __syncthreads();

    // ---- normalize + affine + residual, coalesced store
    #pragma unroll
    for (int i = 0; i < 8; ++i) {
        int e = tid + i * 256;          // 0..2047
        int m = e >> 7;
        int c = e & 127;
        float v = (outS[m][c] - muS[m]) * rsS[m] * lns[c] + lnb[c];
        out[base + (long)m * 128 + c] = nf[base + (long)m * 128 + c] + v;
    }
}

extern "C" void kernel_launch(void* const* d_in, const int* in_sizes, int n_in,
                              void* d_out, int out_size, void* d_ws, size_t ws_size,
                              hipStream_t stream) {
    const float* nf  = (const float*)d_in[0];
    const float* ef  = (const float*)d_in[1];
    const int*   rcv = (const int*)d_in[2];
    const float* W1  = (const float*)d_in[3];
    const float* b1  = (const float*)d_in[4];
    const float* W2  = (const float*)d_in[5];
    const float* b2  = (const float*)d_in[6];
    const float* lns = (const float*)d_in[7];
    const float* lnb = (const float*)d_in[8];
    float* out = (float*)d_out;

    // workspace layout: sums[B*N*128]f32 | cnt[B*N]f32 | W1T bf16 | W2T bf16
    char*  ws   = (char*)d_ws;
    size_t off0 = (size_t)BDIM * NN * 128 * sizeof(float);   // 51.2 MB
    size_t off1 = off0 + (size_t)BDIM * NN * sizeof(float);  // + 400 KB
    float*  sums = (float*)ws;
    float*  cnt  = (float*)(ws + off0);
    __bf16* W1T  = (__bf16*)(ws + off1);
    __bf16* W2T  = (__bf16*)(ws + off1 + (size_t)256 * 128 * 2);

    long nz = (long)BDIM * NN * 129;                 // sums + cnt contiguous
    zero_ws<<<(int)((nz + 255) / 256), 256, 0, stream>>>(sums, nz);
    convert_weights<<<(256 * 128) / 256, 256, 0, stream>>>(W1, W2, W1T, W2T);
    scatter_edges<<<(int)(((long)BDIM * EE * 128) / 256), 256, 0, stream>>>(ef, rcv, sums, cnt);
    node_mlp<<<BDIM * NTILES, 256, 0, stream>>>(nf, sums, cnt, W1T, W2T,
                                                b1, b2, lns, lnb, out);
}
